// BigramLanguageModel_79834852098271
// MI455X (gfx1250) — matrix operand order repaired and checked by reading, not compiled, not run on hardware
//
#include <hip/hip_runtime.h>
#include <hip/hip_bf16.h>

// ---------------------------------------------------------------------------
// Sizes (fixed by the reference)
// ---------------------------------------------------------------------------
#define VOCAB   32000
#define N_EMBD  1024
#define NUM_H   16
#define HS      64
#define BB      4
#define TT      1024
#define BT      (BB * TT)          // 4096 rows

#define AKP 40   // LDS pitch (halves) for K=32 tiles: 80B rows -> conflict-free b128
#define KP  72   // LDS pitch (halves) for 64-wide tiles: 144B rows -> conflict-free

typedef __attribute__((ext_vector_type(16))) _Float16 v16h;
typedef __attribute__((ext_vector_type(8)))  _Float16 v8h;
typedef __attribute__((ext_vector_type(4)))  _Float16 v4h;
typedef __attribute__((ext_vector_type(8)))  float    v8f;

union V8F { v8f v; float f[8]; };

#define WMMA_F16(a, b, c) \
    __builtin_amdgcn_wmma_f32_16x16x32_f16(false, (a), false, (b), (short)0, (c), false, false)

// ---------------------------------------------------------------------------
// Fragment loaders (layouts per CDNA5 ISA 7.12.2, wave32) — all vector loads.
// ---------------------------------------------------------------------------
__device__ inline v16h ldsA(const _Float16* p, int ldm, int lane) {
    int lo = lane & 15, hi = lane >> 4;
    const _Float16* r = p + lo * ldm + hi * 8;
    v8h x0 = *(const v8h*)(r);
    v8h x1 = *(const v8h*)(r + 16);
    v16h a;
#pragma unroll
    for (int i = 0; i < 8; i++) { a[i] = x0[i]; a[i + 8] = x1[i]; }
    return a;
}

// B 32x16 f16 from an n-major (transposed) tile: B[k][n] = p[n*ld + k].
__device__ inline v16h ldsB_nk(const _Float16* p, int ld, int lane) {
    int lo = lane & 15, hi = lane >> 4;
    const _Float16* r = p + lo * ld + hi * 8;
    v8h x0 = *(const v8h*)(r);
    v8h x1 = *(const v8h*)(r + 16);
    v16h b;
#pragma unroll
    for (int i = 0; i < 8; i++) { b[i] = x0[i]; b[i + 8] = x1[i]; }
    return b;
}

// 16-lane butterfly reductions (wave32; xor masks 1..8 stay inside each half).
__device__ inline float rowmax16(float x) {
    x = fmaxf(x, __shfl_xor(x, 1, 32));
    x = fmaxf(x, __shfl_xor(x, 2, 32));
    x = fmaxf(x, __shfl_xor(x, 4, 32));
    x = fmaxf(x, __shfl_xor(x, 8, 32));
    return x;
}
__device__ inline float rowsum16(float x) {
    x += __shfl_xor(x, 1, 32);
    x += __shfl_xor(x, 2, 32);
    x += __shfl_xor(x, 4, 32);
    x += __shfl_xor(x, 8, 32);
    return x;
}

// ---------------------------------------------------------------------------
// K0: tiled transpose + f32->f16.  in: f32 [rows][cols] -> out: f16 [cols][rows].
// ---------------------------------------------------------------------------
__global__ void cvt_transpose(const float* __restrict__ in, _Float16* __restrict__ out,
                              int rows, int cols, long ibstride, long obstride) {
    __shared__ float tile[32][33];
    long ib = (long)blockIdx.z * ibstride;
    long ob = (long)blockIdx.z * obstride;
    int r0 = blockIdx.y * 32, c0 = blockIdx.x * 32;
    int tx = threadIdx.x, ty = threadIdx.y;
#pragma unroll
    for (int j = 0; j < 32; j += 8)
        tile[ty + j][tx] = in[ib + (long)(r0 + ty + j) * cols + c0 + tx];
    __syncthreads();
#pragma unroll
    for (int j = 0; j < 32; j += 8)
        out[ob + (long)(c0 + ty + j) * rows + r0 + tx] = (_Float16)tile[tx][ty + j];
}

// ---------------------------------------------------------------------------
// K1: embedding -> f16 activations
// ---------------------------------------------------------------------------
__global__ void embed_f16(const int* __restrict__ idx, const float* __restrict__ tok,
                          const float* __restrict__ pos, _Float16* __restrict__ xh) {
    int row   = blockIdx.x;
    int t     = row & (TT - 1);
    int token = idx[row];
    const float* te = tok + (long)token * N_EMBD;
    const float* pe = pos + (long)t * N_EMBD;
    _Float16*   orow = xh + (long)row * N_EMBD;
    for (int e = threadIdx.x * 4; e < N_EMBD; e += blockDim.x * 4) {
        float4 a = *(const float4*)(te + e);
        float4 b = *(const float4*)(pe + e);
        v4h o;
        o[0] = (_Float16)(a.x + b.x); o[1] = (_Float16)(a.y + b.y);
        o[2] = (_Float16)(a.z + b.z); o[3] = (_Float16)(a.w + b.w);
        *(v4h*)(orow + e) = o;
    }
}

// ---------------------------------------------------------------------------
// K2: fused QKV projection.  grid (T/64, H, B), 256 threads (8 waves).
// Wave w: one M tile (w>>1), two N tiles ((w&1)*2 + {0,1}) -> A frag loaded
// once per k-step, reused across 6 WMMAs (q/k/v x 2 N tiles).
// ---------------------------------------------------------------------------
__global__ void qkv_gemm(const _Float16* __restrict__ xh,
                         const _Float16* __restrict__ Wqt,
                         const _Float16* __restrict__ Wkt,
                         const _Float16* __restrict__ Wvt,
                         _Float16* __restrict__ qm, _Float16* __restrict__ km,
                         _Float16* __restrict__ vt) {
    int mt = blockIdx.x, h = blockIdx.y, b = blockIdx.z;
    int rb = mt * 64;
    long xrow0 = (long)b * TT + rb;

    __shared__ __align__(16) _Float16 smem[4 * 64 * AKP];   // As + 3x BsT
    _Float16* As  = smem;
    _Float16* BsT[3] = { smem + 64 * AKP, smem + 2 * 64 * AKP, smem + 3 * 64 * AKP };

    int tid = threadIdx.x, lane = tid & 31, w = tid >> 5;
    int lo = lane & 15, hi = lane >> 4;
    int mi = w >> 1, ns = (w & 1) * 2;

    const _Float16* Wp[3] = { Wqt + (long)h * HS * N_EMBD,
                              Wkt + (long)h * HS * N_EMBD,
                              Wvt + (long)h * HS * N_EMBD };
    v8f zf = {};
    V8F acc[3][2];
#pragma unroll
    for (int m = 0; m < 3; m++) { acc[m][0].v = zf; acc[m][1].v = zf; }

    int bn = tid & 63, bkg = (tid >> 6) * 8;

    for (int k0 = 0; k0 < N_EMBD; k0 += 32) {
        __syncthreads();
        {   int i = tid * 8, r = i >> 5, c = i & 31;
            *(v8h*)&As[r * AKP + c] = *(const v8h*)&xh[(xrow0 + r) * N_EMBD + k0 + c];
        }
#pragma unroll
        for (int m = 0; m < 3; m++) {
            *(v8h*)&BsT[m][bn * AKP + bkg] =
                *(const v8h*)&Wp[m][(long)bn * N_EMBD + k0 + bkg];
        }
        __syncthreads();
        v16h a = ldsA(&As[mi * 16 * AKP], AKP, lane);
#pragma unroll
        for (int j = 0; j < 2; j++) {
            int ni = ns + j;
#pragma unroll
            for (int m = 0; m < 3; m++) {
                v16h bf = ldsB_nk(&BsT[m][(ni * 16) * AKP], AKP, lane);
                acc[m][j].v = WMMA_F16(a, bf, acc[m][j].v);
            }
        }
    }

    // q, k: store [b,h,s,d] (f16)
    _Float16* outp[2] = { qm, km };
    long base = ((long)(b * NUM_H + h) * TT + rb);
#pragma unroll
    for (int j = 0; j < 2; j++) {
        int ni = ns + j;
#pragma unroll
        for (int m = 0; m < 2; m++) {
#pragma unroll
            for (int vv = 0; vv < 8; vv++) {
                long r = base + mi * 16 + vv + hi * 8;
                outp[m][r * HS + ni * 16 + lo] = (_Float16)acc[m][j].f[vv];
            }
        }
    }

    // v: transpose 64x64 tile through LDS, then vector stores to [b,h,d,s]
    __syncthreads();
    _Float16* vtile = smem;                     // reuse (64*KP <= 4*64*AKP)
#pragma unroll
    for (int j = 0; j < 2; j++) {
        int ni = ns + j;
#pragma unroll
        for (int vv = 0; vv < 8; vv++)
            vtile[(ni * 16 + lo) * KP + mi * 16 + vv + hi * 8] = (_Float16)acc[2][j].f[vv];
    }
    __syncthreads();
    long hb2 = (long)(b * NUM_H + h) * HS;
#pragma unroll
    for (int u = 0; u < 2; u++) {
        int id = tid + u * 256;
        int d = id >> 3, sg = (id & 7) * 8;
        *(v8h*)&vt[(hb2 + d) * TT + rb + sg] = *(const v8h*)&vtile[d * KP + sg];
    }
}

// ---------------------------------------------------------------------------
// K3: flash attention.  grid (T/64, H, B), 128 threads (4 waves).
// ---------------------------------------------------------------------------
__global__ void flash_attn(const _Float16* __restrict__ qm,
                           const _Float16* __restrict__ km,
                           const _Float16* __restrict__ vt,
                           _Float16* __restrict__ att_out) {
    int mt = blockIdx.x, h = blockIdx.y, b = blockIdx.z;
    int rb = mt * 64;
    long hb  = (long)(b * NUM_H + h) * TT;
    long hb2 = (long)(b * NUM_H + h) * HS;

    __shared__ __align__(16) _Float16 Qs[64 * KP];
    __shared__ __align__(16) _Float16 Ks[64 * KP];
    __shared__ __align__(16) _Float16 VsT[64 * KP];
    __shared__ __align__(16) _Float16 Ps[64 * KP];

    int tid = threadIdx.x, lane = tid & 31, w = tid >> 5;
    int lo = lane & 15, hi = lane >> 4;

#pragma unroll
    for (int u = 0; u < 4; u++) {
        int i = (tid + u * 128) * 8, r = i >> 6, c = i & 63;
        *(v8h*)&Qs[r * KP + c] = *(const v8h*)&qm[(hb + rb + r) * HS + c];
    }

    v8f zf = {};
    V8F O[4];
#pragma unroll
    for (int di = 0; di < 4; di++) O[di].v = zf;
    float mrun[8], lrun[8];
#pragma unroll
    for (int vv = 0; vv < 8; vv++) { mrun[vv] = -1e30f; lrun[vv] = 0.f; }

    for (int kb = 0; kb <= mt; kb++) {
        int cs = kb * 64;
        __syncthreads();
#pragma unroll
        for (int u = 0; u < 4; u++) {
            int i = (tid + u * 128) * 8, r = i >> 6, c = i & 63;
            *(v8h*)&Ks[r * KP + c] = *(const v8h*)&km[(hb + cs + r) * HS + c];
            int id = tid + u * 128;
            int d = id & 63, sg = (id >> 6) * 8;
            *(v8h*)&VsT[d * KP + sg] = *(const v8h*)&vt[(hb2 + d) * TT + cs + sg];
        }
        __syncthreads();

        V8F S[4];
#pragma unroll
        for (int ci = 0; ci < 4; ci++) S[ci].v = zf;
#pragma unroll
        for (int c2 = 0; c2 < 2; c2++) {
            v16h a = ldsA(&Qs[(w * 16) * KP + c2 * 32], KP, lane);
#pragma unroll
            for (int ci = 0; ci < 4; ci++) {
                v16h bf = ldsB_nk(&Ks[(ci * 16) * KP + c2 * 32], KP, lane);
                S[ci].v = WMMA_F16(a, bf, S[ci].v);
            }
        }

        float sarr[4][8];
#pragma unroll
        for (int ci = 0; ci < 4; ci++)
#pragma unroll
            for (int vv = 0; vv < 8; vv++) sarr[ci][vv] = S[ci].f[vv];

        if (kb == mt) {
#pragma unroll
            for (int ci = 0; ci < 4; ci++)
#pragma unroll
                for (int vv = 0; vv < 8; vv++) {
                    int colg = cs + ci * 16 + lo;
                    int rowg = rb + w * 16 + vv + hi * 8;
                    if (colg > rowg) sarr[ci][vv] = -1e30f;
                }
        }

#pragma unroll
        for (int vv = 0; vv < 8; vv++) {
            float rm = fmaxf(fmaxf(sarr[0][vv], sarr[1][vv]),
                             fmaxf(sarr[2][vv], sarr[3][vv]));
            rm = rowmax16(rm);
            float mn = fmaxf(mrun[vv], rm);
            float sc = __expf(mrun[vv] - mn);
            mrun[vv] = mn;
            lrun[vv] *= sc;
#pragma unroll
            for (int di = 0; di < 4; di++) O[di].f[vv] *= sc;
            float ps = 0.f;
#pragma unroll
            for (int ci = 0; ci < 4; ci++) {
                float pe = __expf(sarr[ci][vv] - mn);
                ps += pe;
                Ps[(w * 16 + vv + hi * 8) * KP + ci * 16 + lo] = (_Float16)pe;
            }
            lrun[vv] += rowsum16(ps);
        }

#pragma unroll
        for (int c2 = 0; c2 < 2; c2++) {
            v16h a = ldsA(&Ps[(w * 16) * KP + c2 * 32], KP, lane);
#pragma unroll
            for (int di = 0; di < 4; di++) {
                v16h bf = ldsB_nk(&VsT[(di * 16) * KP + c2 * 32], KP, lane);
                O[di].v = WMMA_F16(a, bf, O[di].v);
            }
        }
    }

#pragma unroll
    for (int vv = 0; vv < 8; vv++) {
        float inv = 1.f / lrun[vv];
        long row = (long)b * TT + rb + w * 16 + vv + hi * 8;
#pragma unroll
        for (int di = 0; di < 4; di++) {
            att_out[row * (NUM_H * HS) + h * HS + di * 16 + lo] =
                (_Float16)(O[di].f[vv] * inv);
        }
    }
}

// ---------------------------------------------------------------------------
// K4: lm_head GEMM  [4096,1024]f16 x W^T[32000,1024]f16 + bias -> f32 logits.
// 128x128 block tile, 256 threads; wave = 2M x 4N tiles (8 WMMA per k-step,
// 12 LDS loads -> 1.5 ds/WMMA).  Double-buffered LDS staging (static buffers)
// -> one barrier per k-step, global loads for k+1 overlap WMMAs for k.
// W^T (64 MB f16) is L2-resident on MI455X (192 MB L2).
// ---------------------------------------------------------------------------
__device__ inline void lm_stage(const _Float16* __restrict__ A,
                                const _Float16* __restrict__ Wt,
                                _Float16* As, _Float16* Bs,
                                int mb, int nb, int k0, int tid) {
#pragma unroll
    for (int u = 0; u < 2; u++) {
        int id = tid + u * 256;                 // 512 v8h tasks per tile
        int r = id >> 2, c = (id & 3) * 8;
        *(v8h*)&As[r * AKP + c] = *(const v8h*)&A[(long)(mb + r) * N_EMBD + k0 + c];
        const _Float16* src = &Wt[(long)(nb + r) * N_EMBD + k0 + c];
        *(v8h*)&Bs[r * AKP + c] = *(const v8h*)src;
        if (k0 + 64 < N_EMBD)                   // global_prefetch_b8, k+2 tile
            __builtin_prefetch(src + 64, 0, 0);
    }
}

__device__ inline void lm_compute(const _Float16* As, const _Float16* Bs,
                                  V8F acc[2][4], int mi0, int ns, int lane) {
    v16h a0 = ldsA(&As[(mi0 + 0) * 16 * AKP], AKP, lane);
    v16h a1 = ldsA(&As[(mi0 + 1) * 16 * AKP], AKP, lane);
#pragma unroll
    for (int ni = 0; ni < 4; ni++) {
        v16h bf = ldsB_nk(&Bs[(ns + ni) * 16 * AKP], AKP, lane);
        acc[0][ni].v = WMMA_F16(a0, bf, acc[0][ni].v);
        acc[1][ni].v = WMMA_F16(a1, bf, acc[1][ni].v);
    }
}

__global__ void lm_head(const _Float16* __restrict__ A,
                        const _Float16* __restrict__ Wt,   // [VOCAB][N_EMBD]
                        const float* __restrict__ bias,
                        float* __restrict__ out) {
    const int N = VOCAB;
    int nb = blockIdx.x * 128, mb = blockIdx.y * 128;

    __shared__ __align__(16) _Float16 As0[128 * AKP];
    __shared__ __align__(16) _Float16 As1[128 * AKP];
    __shared__ __align__(16) _Float16 Bs0[128 * AKP];
    __shared__ __align__(16) _Float16 Bs1[128 * AKP];

    int tid = threadIdx.x, lane = tid & 31, w = tid >> 5;
    int mi0 = (w >> 1) * 2, ns = (w & 1) * 4;

    v8f zf = {};
    V8F acc[2][4];
#pragma unroll
    for (int mj = 0; mj < 2; mj++)
#pragma unroll
        for (int ni = 0; ni < 4; ni++) acc[mj][ni].v = zf;

    lm_stage(A, Wt, As0, Bs0, mb, nb, 0, tid);
    for (int kt = 0; kt < N_EMBD / 32; kt += 2) {
        __syncthreads();
        if ((kt + 1) * 32 < N_EMBD)
            lm_stage(A, Wt, As1, Bs1, mb, nb, (kt + 1) * 32, tid);
        lm_compute(As0, Bs0, acc, mi0, ns, lane);
        __syncthreads();
        if ((kt + 2) * 32 < N_EMBD)
            lm_stage(A, Wt, As0, Bs0, mb, nb, (kt + 2) * 32, tid);
        lm_compute(As1, Bs1, acc, mi0, ns, lane);
    }

    int lo = lane & 15, hi = lane >> 4;
#pragma unroll
    for (int mj = 0; mj < 2; mj++) {
#pragma unroll
        for (int ni = 0; ni < 4; ni++) {
#pragma unroll
            for (int vv = 0; vv < 8; vv++) {
                int r = mb + (mi0 + mj) * 16 + vv + hi * 8;
                int c = nb + (ns + ni) * 16 + lo;
                out[(long)r * N + c] = acc[mj][ni].f[vv] + bias[c];
            }
        }
    }
}

// ---------------------------------------------------------------------------
// Host launcher
// ---------------------------------------------------------------------------
extern "C" void kernel_launch(void* const* d_in, const int* in_sizes, int n_in,
                              void* d_out, int out_size, void* d_ws, size_t ws_size,
                              hipStream_t stream) {
    (void)in_sizes; (void)n_in; (void)out_size; (void)ws_size;
    const int*   idx = (const int*)  d_in[0];
    const float* tok = (const float*)d_in[1];
    const float* pos = (const float*)d_in[2];
    const float* Wq  = (const float*)d_in[3];
    const float* Wk  = (const float*)d_in[4];
    const float* Wv  = (const float*)d_in[5];
    const float* lmW = (const float*)d_in[6];
    const float* lmb = (const float*)d_in[7];
    float* out = (float*)d_out;

    char* ws = (char*)d_ws;
    size_t off = 0;
    auto take = [&](size_t bytes) -> void* {
        void* p = ws + off;
        off += (bytes + 255) & ~(size_t)255;
        return p;
    };
    const size_t nW   = (size_t)NUM_H * N_EMBD * HS;
    const size_t nLW  = (size_t)N_EMBD * VOCAB;
    const size_t nQKV = (size_t)BB * NUM_H * TT * HS;

    _Float16* xh   = (_Float16*)take((size_t)BT * N_EMBD * 2);
    _Float16* Wqt  = (_Float16*)take(nW * 2);
    _Float16* Wkt  = (_Float16*)take(nW * 2);
    _Float16* Wvt  = (_Float16*)take(nW * 2);
    _Float16* lmWt = (_Float16*)take(nLW * 2);
    _Float16* qm   = (_Float16*)take(nQKV * 2);
    _Float16* km   = (_Float16*)take(nQKV * 2);
    _Float16* vt   = (_Float16*)take(nQKV * 2);
    _Float16* att  = (_Float16*)take((size_t)BT * N_EMBD * 2);

    dim3 tb(32, 8);
    long ws1 = (long)N_EMBD * HS;
    cvt_transpose<<<dim3(HS / 32, N_EMBD / 32, NUM_H), tb, 0, stream>>>(Wq, Wqt, N_EMBD, HS, ws1, ws1);
    cvt_transpose<<<dim3(HS / 32, N_EMBD / 32, NUM_H), tb, 0, stream>>>(Wk, Wkt, N_EMBD, HS, ws1, ws1);
    cvt_transpose<<<dim3(HS / 32, N_EMBD / 32, NUM_H), tb, 0, stream>>>(Wv, Wvt, N_EMBD, HS, ws1, ws1);
    cvt_transpose<<<dim3(VOCAB / 32, N_EMBD / 32, 1), tb, 0, stream>>>(lmW, lmWt, N_EMBD, VOCAB, 0, 0);

    embed_f16<<<BT, 256, 0, stream>>>(idx, tok, pos, xh);

    qkv_gemm<<<dim3(TT / 64, NUM_H, BB), 256, 0, stream>>>(xh, Wqt, Wkt, Wvt, qm, km, vt);

    flash_attn<<<dim3(TT / 64, NUM_H, BB), 128, 0, stream>>>(qm, km, vt, att);

    lm_head<<<dim3(VOCAB / 128, BT / 128), 256, 0, stream>>>(att, lmWt, lmb, out);
}
